// FourierWindowAttention_43130061586464
// MI455X (gfx1250) — matrix operand order repaired and checked by reading, not compiled, not run on hardware
//
#include <hip/hip_runtime.h>
#include <hip/hip_bf16.h>

typedef __attribute__((ext_vector_type(16))) _Float16 v16h;
typedef __attribute__((ext_vector_type(8)))  float    v8f;
typedef __attribute__((ext_vector_type(4)))  unsigned int u32x4;
typedef __attribute__((ext_vector_type(8)))  int      i32x8;
typedef __attribute__((ext_vector_type(4)))  int      i32x4;

#define DIM    192
#define HEADS  6
#define HD     32
#define NTOK   64
#define NWIN   64
#define NFREQ  66      // real spectral token count: 2*(64/2+1)
#define NFP    96      // padded spectral token count (3 K-steps of 32)
#define LDA    200     // f16 row stride for 192-col token buffers
#define LDQ    592     // f16 row stride for 576-col qkv buffers
#define SCALE  0.1767766952966369f   // 1/sqrt(32)
#define TWO_PI 6.283185307179586f

// ---- byte offsets into the dynamic LDS arena (all 16B aligned) ----
#define OFF_XH    0          // 64  x LDA  f16 : x (f16), later x_sp (f16) for proj
#define OFF_XF    25600      // NFP x LDA  f16 : xf (stacked rfft), later xff
#define OFF_QKV   64000      // NFP x LDQ  f16 : qkv (spatial rows 0..63, then spectral)
#define OFF_XSP   177664     // 64  x 192  f32 : TDM staging for x, then token accumulator
#define OFF_FM    226816     // NFP x NFP  f16 : DFT forward / inverse matrix
#define OFF_S     245248     // NFP x NFP  f32 : attention scores
#define OFF_P     282112     // NFP x NFP  f16 : softmaxed probabilities
#define SMEM_BYTES 300544

// Tensor Data Mover availability (guarded so both toolchains compile).
#if defined(__AMDGCN__) && defined(__gfx1250__) && \
    __has_builtin(__builtin_amdgcn_tensor_load_to_lds) && \
    __has_builtin(__builtin_amdgcn_s_wait_tensorcnt)
#define HAS_TDM 1
#endif

// -------- WMMA fragment helpers (wave32; A/B k order is contiguous, k = 16*(lane/16) + i) --------

// A fragment, 16x32 f16: lanes 0-15 -> M=lane, K=0..15;
// lanes 16-31 -> M=lane-16, K=16..31.
static __device__ inline v16h frag_a(const _Float16* a0, int lda) {
  const int lane = threadIdx.x & 31;
  const int m  = lane & 15;
  const int kb = (lane >> 4) << 4;          // 0 or 16
  const _Float16* r = a0 + m * lda + kb;
  v16h f;
#pragma unroll
  for (int i = 0; i < 16; ++i) f[i] = r[i];
  return f;
}

// B fragment, 32x16 f16, stored K-major (K rows x N cols, row stride ldb):
// lanes 0-15 -> N=lane, K=0..15; lanes 16-31 -> N=lane-16, K=16..31.
static __device__ inline v16h frag_b_k(const _Float16* b0, int ldb) {
  const int lane = threadIdx.x & 31;
  const int n  = lane & 15;
  const int kb = (lane >> 4) << 4;          // 0 or 16
  v16h f;
#pragma unroll
  for (int i = 0; i < 16; ++i) f[i] = b0[(kb + i) * ldb + n];
  return f;
}

// B fragment from an N-major (transposed) f16 buffer: element (k,n) = b0[n*ldb + k].
static __device__ inline v16h frag_b_n(const _Float16* b0, int ldb) {
  const int lane = threadIdx.x & 31;
  const int n  = lane & 15;
  const int kb = (lane >> 4) << 4;
  const _Float16* r = b0 + n * ldb + kb;
  v16h f;
#pragma unroll
  for (int i = 0; i < 16; ++i) f[i] = r[i];
  return f;
}

// B fragment from a global f32 weight matrix stored N-major (W[n][k]), i.e. B = W^T.
static __device__ inline v16h frag_b_gw(const float* b0, int ldb) {
  const int lane = threadIdx.x & 31;
  const int n  = lane & 15;
  const int kb = (lane >> 4) << 4;
  const float* r = b0 + n * ldb + kb;
  v16h f;
#pragma unroll
  for (int i = 0; i < 16; ++i) f[i] = (_Float16)r[i];
  return f;
}

static __device__ inline v8f wmma_f16(v16h a, v16h b, v8f c) {
  return __builtin_amdgcn_wmma_f32_16x16x32_f16(false, a, false, b, (short)0, c,
                                                false, false);
}

// C/D 16x16 f32: lanes 0-15 -> N=lane, VGPR r -> M=r; lanes 16-31 -> M=r+8.
static __device__ inline void store_c_f32(float* c0, int ldc, v8f acc) {
  const int lane = threadIdx.x & 31;
  const int n  = lane & 15;
  const int mb = (lane >> 4) << 3;
#pragma unroll
  for (int r = 0; r < 8; ++r) c0[(mb + r) * ldc + n] = acc[r];
}
static __device__ inline void addstore_c_f32(float* c0, int ldc, v8f acc) {
  const int lane = threadIdx.x & 31;
  const int n  = lane & 15;
  const int mb = (lane >> 4) << 3;
#pragma unroll
  for (int r = 0; r < 8; ++r) c0[(mb + r) * ldc + n] += acc[r];
}
static __device__ inline void store_c_f16(_Float16* c0, int ldc, v8f acc) {
  const int lane = threadIdx.x & 31;
  const int n  = lane & 15;
  const int mb = (lane >> 4) << 3;
#pragma unroll
  for (int r = 0; r < 8; ++r) c0[(mb + r) * ldc + n] = (_Float16)acc[r];
}
static __device__ inline void store_c_f16_bias(_Float16* c0, int ldc, v8f acc, float bias) {
  const int lane = threadIdx.x & 31;
  const int n  = lane & 15;
  const int mb = (lane >> 4) << 3;
#pragma unroll
  for (int r = 0; r < 8; ++r) c0[(mb + r) * ldc + n] = (_Float16)(acc[r] + bias);
}
static __device__ inline void store_c_gf32_bias(float* c0, int ldc, v8f acc, float bias) {
  const int lane = threadIdx.x & 31;
  const int n  = lane & 15;
  const int mb = (lane >> 4) << 3;
#pragma unroll
  for (int r = 0; r < 8; ++r) c0[(mb + r) * ldc + n] = acc[r] + bias;
}

// -------------------------- fused kernel --------------------------
__global__ __launch_bounds__(256) void fourier_window_attention_kernel(
    const float* __restrict__ xg,    const float* __restrict__ maskg,
    const float* __restrict__ qkvw,  const float* __restrict__ qkvb,
    const float* __restrict__ projw, const float* __restrict__ projb,
    const float* __restrict__ relb,  const int* __restrict__ relidx,
    float* __restrict__ outg, float* __restrict__ attng) {
  extern __shared__ char smem[];
  _Float16* sXH  = (_Float16*)(smem + OFF_XH);
  _Float16* sXF  = (_Float16*)(smem + OFF_XF);    // xf, later xff (aliased)
  _Float16* sQKV = (_Float16*)(smem + OFF_QKV);
  float*    sXSP = (float*)   (smem + OFF_XSP);
  _Float16* sFM  = (_Float16*)(smem + OFF_FM);
  float*    sS   = (float*)   (smem + OFF_S);
  _Float16* sP   = (_Float16*)(smem + OFF_P);

  const int tid  = threadIdx.x;
  const int wave = tid >> 5;
  const int blk  = blockIdx.x;
  const int widx = blk & (NWIN - 1);              // window index within image

  // L2 warmup hints for the shared weight matrices (global_prefetch_b8).
  __builtin_prefetch(qkvw + tid, 0, 1);
  __builtin_prefetch(projw + tid, 0, 1);

  // ---- stage 1: x tile -> LDS ----
  const float* xp = xg + (size_t)blk * NTOK * DIM;
#if defined(HAS_TDM)
  // Tensor Data Mover: DMA the 64x192 f32 tile into the (currently free) sXSP
  // region, then convert to f16. D# per ISA ch.8: group0 = {count, lds_addr,
  // global_addr, type=2}; group1 = {data_size=4B, dims/tile/stride}.
  if (wave == 0) {
    // LDS aperture maps flat addr[31:0] directly to the LDS byte offset.
    const unsigned lds_addr = (unsigned)(unsigned long long)(void*)sXSP;
    const unsigned long long ga = (unsigned long long)(const void*)xp;
    u32x4 g0 = {0u, 0u, 0u, 0u};
    g0[0] = 1u;                                        // count=1 (valid user D#)
    g0[1] = lds_addr;                                  // lds_addr [63:32]
    g0[2] = (unsigned)(ga & 0xffffffffu);              // global_addr lo
    g0[3] = (unsigned)((ga >> 32) & 0x01ffffffu)       // global_addr hi [120:64]
          | (2u << 30);                                // type=2 ("image") [127:126]
    i32x8 g1 = {0, 0, 0, 0, 0, 0, 0, 0};
    g1[0] = (int)(2u << 16);                           // data_size=4B [17:16]
    g1[1] = (int)((unsigned)DIM << 16);                // tensor_dim0=192 [79:48] lo
    g1[2] = (int)((unsigned)NTOK << 16);               // tensor_dim1=64  [111:80] lo
    g1[3] = (int)((unsigned)DIM << 16);                // tile_dim0=192   [127:112]
    g1[4] = NTOK;                                      // tile_dim1=64    [143:128]
    g1[5] = DIM;                                       // tensor_dim0_stride=192 [207:160]
    i32x4 g2 = {0, 0, 0, 0};
    i32x4 g3 = {0, 0, 0, 0};
#if __clang_major__ >= 23
    i32x8 g4 = {0, 0, 0, 0, 0, 0, 0, 0};
    __builtin_amdgcn_tensor_load_to_lds(g0, g1, g2, g3, g4, 0);
#else
    __builtin_amdgcn_tensor_load_to_lds(g0, g1, g2, g3, 0);
#endif
    __builtin_amdgcn_s_wait_tensorcnt(0);
  }
  __syncthreads();
  for (int i = tid; i < NTOK * DIM; i += 256)
    sXH[(i / DIM) * LDA + (i % DIM)] = (_Float16)sXSP[i];
#else
  for (int i = tid; i < NTOK * DIM; i += 256)
    sXH[(i / DIM) * LDA + (i % DIM)] = (_Float16)xp[i];
#endif
  __syncthreads();

  // ---- stage 2: spatial qkv = x @ Wqkv^T + b   (64 x 576, K=192) ----
  for (int t = wave; t < 4 * 36; t += 8) {
    const int mt = t / 36, nt = t % 36;
    v8f acc = {};
#pragma unroll
    for (int kk = 0; kk < DIM; kk += 32)
      acc = wmma_f16(frag_a(sXH + mt * 16 * LDA + kk, LDA),
                     frag_b_gw(qkvw + (nt * 16) * DIM + kk, DIM), acc);
    store_c_f16_bias(sQKV + mt * 16 * LDQ + nt * 16, LDQ, acc,
                     qkvb[nt * 16 + (tid & 15)]);
  }
  __syncthreads();

  // ---- stage 3: spatial attention per head ----
  for (int h = 0; h < HEADS; ++h) {
    // scores S = q @ k^T : 4x4 tiles, K = 32 (one WMMA each)
    for (int t = wave; t < 16; t += 8) {
      const int mt = t >> 2, nt = t & 3;
      v8f acc = {};
      acc = wmma_f16(frag_a(sQKV + mt * 16 * LDQ + h * HD, LDQ),
                     frag_b_n(sQKV + nt * 16 * LDQ + DIM + h * HD, LDQ), acc);
      store_c_f32(sS + mt * 16 * NFP + nt * 16, NFP, acc);
    }
    __syncthreads();
    // fused scale + rel-pos bias + window mask + softmax; emit attn output
    if (tid < NTOK) {
      const int r = tid;
      float* srow = sS + r * NFP;
      const int* irow = relidx + r * NTOK;
      const float* mrow = maskg + (size_t)widx * NTOK * NTOK + r * NTOK;
      float mx = -3.0e38f;
      for (int m = 0; m < NTOK; ++m) {
        float v = srow[m] * SCALE + relb[irow[m] * HEADS + h] + mrow[m];
        srow[m] = v;
        mx = fmaxf(mx, v);
      }
      float sum = 0.f;
      for (int m = 0; m < NTOK; ++m) {
        float e = __expf(srow[m] - mx);
        srow[m] = e;
        sum += e;
      }
      const float inv = 1.f / sum;
      float* ag = attng + (((size_t)blk * HEADS + h) * NTOK + r) * NTOK;
      _Float16* prow = sP + r * NFP;
      for (int m = 0; m < NTOK; ++m) {
        const float p = srow[m] * inv;
        ag[m] = p;
        prow[m] = (_Float16)p;
      }
    }
    __syncthreads();
    // x_sp[:, h*32:(h+1)*32] = P @ v : 4x2 tiles, K = 64
    for (int t = wave; t < 8; t += 8) {
      const int mt = t >> 1, nt = t & 1;
      v8f acc = {};
#pragma unroll
      for (int kk = 0; kk < NTOK; kk += 32)
        acc = wmma_f16(frag_a(sP + mt * 16 * NFP + kk, NFP),
                       frag_b_k(sQKV + kk * LDQ + 2 * DIM + h * HD + nt * 16, LDQ),
                       acc);
      store_c_f32(sXSP + mt * 16 * DIM + h * HD + nt * 16, DIM, acc);
    }
    __syncthreads();
  }

  // ---- stage 4: forward real-DFT (ortho) as a GEMM: xf = F @ x ----
  // F is (NFP x 64) f16: rows 0..32 cos/8, rows 33..65 -sin/8, rest zero.
  for (int i = tid; i < NFP * NFP; i += 256) {
    const int f = i / NFP, n = i % NFP;
    float v = 0.f;
    if (n < NTOK) {
      if (f < 33)       v =  __cosf(TWO_PI * (float)(f * n) / 64.f) * 0.125f;
      else if (f < 66)  v = -__sinf(TWO_PI * (float)((f - 33) * n) / 64.f) * 0.125f;
    }
    sFM[f * NFP + n] = (_Float16)v;
  }
  __syncthreads();
  for (int t = wave; t < 6 * 12; t += 8) {       // 96x192, K=64
    const int mt = t / 12, nt = t % 12;
    v8f acc = {};
#pragma unroll
    for (int kk = 0; kk < NTOK; kk += 32)
      acc = wmma_f16(frag_a(sFM + mt * 16 * NFP + kk, NFP),
                     frag_b_k(sXH + kk * LDA + nt * 16, LDA), acc);
    store_c_f16(sXF + mt * 16 * LDA + nt * 16, LDA, acc);
  }
  __syncthreads();

  // ---- stage 5: spectral qkv = xf @ Wqkv^T + b  (96 x 576, K=192) ----
  for (int t = wave; t < 6 * 36; t += 8) {
    const int mt = t / 36, nt = t % 36;
    v8f acc = {};
#pragma unroll
    for (int kk = 0; kk < DIM; kk += 32)
      acc = wmma_f16(frag_a(sXF + mt * 16 * LDA + kk, LDA),
                     frag_b_gw(qkvw + (nt * 16) * DIM + kk, DIM), acc);
    store_c_f16_bias(sQKV + mt * 16 * LDQ + nt * 16, LDQ, acc,
                     qkvb[nt * 16 + (tid & 15)]);
  }
  __syncthreads();

  // ---- stage 6: spectral attention per head (66 valid tokens, padded to 96) ----
  for (int h = 0; h < HEADS; ++h) {
    for (int t = wave; t < 36; t += 8) {         // 6x6 score tiles, K=32
      const int mt = t / 6, nt = t % 6;
      v8f acc = {};
      acc = wmma_f16(frag_a(sQKV + mt * 16 * LDQ + h * HD, LDQ),
                     frag_b_n(sQKV + nt * 16 * LDQ + DIM + h * HD, LDQ), acc);
      store_c_f32(sS + mt * 16 * NFP + nt * 16, NFP, acc);
    }
    __syncthreads();
    if (tid < NFP) {
      const int r = tid;
      _Float16* prow = sP + r * NFP;
      if (r < NFREQ) {
        float* srow = sS + r * NFP;
        float mx = -3.0e38f;
        for (int m = 0; m < NFREQ; ++m) {
          float v = srow[m] * SCALE;
          srow[m] = v;
          mx = fmaxf(mx, v);
        }
        float sum = 0.f;
        for (int m = 0; m < NFREQ; ++m) {
          float e = __expf(srow[m] - mx);
          srow[m] = e;
          sum += e;
        }
        const float inv = 1.f / sum;
        for (int m = 0; m < NFREQ; ++m) prow[m] = (_Float16)(srow[m] * inv);
        for (int m = NFREQ; m < NFP; ++m) prow[m] = (_Float16)0.f;
      } else {
        for (int m = 0; m < NFP; ++m) prow[m] = (_Float16)0.f;  // zero pad rows
      }
    }
    __syncthreads();
    // xff[:, h*32:(h+1)*32] = P @ v : 6x2 tiles, K = 96 (pads are exact zeros)
    for (int t = wave; t < 12; t += 8) {
      const int mt = t >> 1, nt = t & 1;
      v8f acc = {};
#pragma unroll
      for (int kk = 0; kk < NFP; kk += 32)
        acc = wmma_f16(frag_a(sP + mt * 16 * NFP + kk, NFP),
                       frag_b_k(sQKV + kk * LDQ + 2 * DIM + h * HD + nt * 16, LDQ),
                       acc);
      store_c_f16(sXF + mt * 16 * LDA + h * HD + nt * 16, LDA, acc);  // xff over xf
    }
    __syncthreads();
  }

  // ---- stage 7: inverse real-DFT (ortho) matrix iF (64 x NFP) ----
  // x[n] = (1/8)[Xr0 + (-1)^n Xr32 + sum_{f=1..31} 2(Xr_f cos - Xi_f sin)]
  for (int i = tid; i < NTOK * NFP; i += 256) {
    const int n = i / NFP, f = i % NFP;
    float v = 0.f;
    if (f == 0)                 v = 0.125f;
    else if (f == 32)           v = (n & 1) ? -0.125f : 0.125f;
    else if (f < 32)            v =  0.25f * __cosf(TWO_PI * (float)(f * n) / 64.f);
    else if (f >= 34 && f < 65) v = -0.25f * __sinf(TWO_PI * (float)((f - 33) * n) / 64.f);
    sFM[n * NFP + f] = (_Float16)v;
  }
  __syncthreads();

  // ---- stage 8: x_sp += iF @ xff  (64 x 192, K = 96) ----
  for (int t = wave; t < 4 * 12; t += 8) {
    const int mt = t / 12, nt = t % 12;
    v8f acc = {};
#pragma unroll
    for (int kk = 0; kk < NFP; kk += 32)
      acc = wmma_f16(frag_a(sFM + mt * 16 * NFP + kk, NFP),
                     frag_b_k(sXF + kk * LDA + nt * 16, LDA), acc);
    addstore_c_f32(sXSP + mt * 16 * DIM + nt * 16, DIM, acc);
  }
  __syncthreads();

  // ---- stage 9: out = x_sp @ Wproj^T + b  (64 x 192, K = 192) ----
  for (int i = tid; i < NTOK * DIM; i += 256)    // x_sp -> f16 (reuse sXH)
    sXH[(i / DIM) * LDA + (i % DIM)] = (_Float16)sXSP[i];
  __syncthreads();
  {
    float* op = outg + (size_t)blk * NTOK * DIM;
    for (int t = wave; t < 4 * 12; t += 8) {
      const int mt = t / 12, nt = t % 12;
      v8f acc = {};
#pragma unroll
      for (int kk = 0; kk < DIM; kk += 32)
        acc = wmma_f16(frag_a(sXH + mt * 16 * LDA + kk, LDA),
                       frag_b_gw(projw + (nt * 16) * DIM + kk, DIM), acc);
      store_c_gf32_bias(op + mt * 16 * DIM + nt * 16, DIM, acc,
                        projb[nt * 16 + (tid & 15)]);
    }
  }
}

extern "C" void kernel_launch(void* const* d_in, const int* in_sizes, int n_in,
                              void* d_out, int out_size, void* d_ws, size_t ws_size,
                              hipStream_t stream) {
  (void)in_sizes; (void)n_in; (void)out_size; (void)d_ws; (void)ws_size;
  const float* x      = (const float*)d_in[0];
  const float* mask   = (const float*)d_in[1];
  const float* qkv_w  = (const float*)d_in[2];
  const float* qkv_b  = (const float*)d_in[3];
  const float* proj_w = (const float*)d_in[4];
  const float* proj_b = (const float*)d_in[5];
  const float* relb   = (const float*)d_in[6];
  const int*   relidx = (const int*)d_in[7];

  float* out  = (float*)d_out;
  float* attn = out + (size_t)2048 * NTOK * DIM;   // (out, attn) concatenated

  fourier_window_attention_kernel<<<dim3(2048), dim3(256), SMEM_BYTES, stream>>>(
      x, mask, qkv_w, qkv_b, proj_w, proj_b, relb, relidx, out, attn);
}
